// SAModule_80272938762722
// MI455X (gfx1250) — matrix operand order repaired and checked by reading, not compiled, not run on hardware
//
#include <hip/hip_runtime.h>
#include <hip/hip_bf16.h>
#include <math.h>

// ---------------- problem constants (from reference) ----------------
#define Bc   2
#define Nc   16384
#define Cc   3
#define Mc   2048
#define Kc   64
#define R2c  0.04f          // R*R, R=0.2
#define H1c  64
#define H2c  64
#define HOc  128

typedef __attribute__((ext_vector_type(16))) _Float16 v16h;
typedef __attribute__((ext_vector_type(8)))  float    v8f;

// =====================================================================
// Kernel 1: farthest point sampling. one workgroup per cloud.
// 1024 threads, 16 points per thread kept in registers.
// =====================================================================
#define FPS_PPT 16
__global__ __launch_bounds__(1024) void fps_kernel(
    const float* __restrict__ pos, int* __restrict__ idx,
    float* __restrict__ centers)
{
    __shared__ float rv[32];
    __shared__ int   ri[32];
    __shared__ float bc[3];

    const int b    = blockIdx.x;
    const int t    = threadIdx.x;
    const int lane = t & 31;
    const int wave = t >> 5;
    const float* __restrict__ P = pos + (size_t)b * Nc * 3;

    if (t == 0) {
        bc[0] = P[0]; bc[1] = P[1]; bc[2] = P[2];
        idx[b * Mc] = 0;
        centers[(size_t)(b * Mc) * 3 + 0] = P[0];
        centers[(size_t)(b * Mc) * 3 + 1] = P[1];
        centers[(size_t)(b * Mc) * 3 + 2] = P[2];
    }
    __syncthreads();
    float cx = bc[0], cy = bc[1], cz = bc[2];

    float px[FPS_PPT], py[FPS_PPT], pz[FPS_PPT], md[FPS_PPT];
#pragma unroll
    for (int k = 0; k < FPS_PPT; ++k) {
        const int p = t + k * 1024;
        px[k] = P[p * 3 + 0];
        py[k] = P[p * 3 + 1];
        pz[k] = P[p * 3 + 2];
        const float dx = px[k] - cx, dy = py[k] - cy, dz = pz[k] - cz;
        md[k] = dx * dx + dy * dy + dz * dz;
    }

    for (int it = 1; it < Mc; ++it) {
        // local argmax (tie -> lowest index, like jnp.argmax)
        float bv = -1.0f; int bi = 0x7fffffff;
#pragma unroll
        for (int k = 0; k < FPS_PPT; ++k) {
            const int p = t + k * 1024;
            if (md[k] > bv) { bv = md[k]; bi = p; }
        }
        // wave reduction
#pragma unroll
        for (int off = 16; off > 0; off >>= 1) {
            const float ov = __shfl_xor(bv, off);
            const int   oi = __shfl_xor(bi, off);
            if (ov > bv || (ov == bv && oi < bi)) { bv = ov; bi = oi; }
        }
        if (lane == 0) { rv[wave] = bv; ri[wave] = bi; }
        __syncthreads();
        if (wave == 0) {
            float v2 = rv[lane]; int i2 = ri[lane];
#pragma unroll
            for (int off = 16; off > 0; off >>= 1) {
                const float ov = __shfl_xor(v2, off);
                const int   oi = __shfl_xor(i2, off);
                if (ov > v2 || (ov == v2 && oi < i2)) { v2 = ov; i2 = oi; }
            }
            if (lane == 0) {
                const float qx = P[i2 * 3 + 0];
                const float qy = P[i2 * 3 + 1];
                const float qz = P[i2 * 3 + 2];
                bc[0] = qx; bc[1] = qy; bc[2] = qz;
                idx[b * Mc + it] = i2;
                const size_t co = (size_t)(b * Mc + it) * 3;
                centers[co + 0] = qx; centers[co + 1] = qy; centers[co + 2] = qz;
            }
        }
        __syncthreads();
        cx = bc[0]; cy = bc[1]; cz = bc[2];
#pragma unroll
        for (int k = 0; k < FPS_PPT; ++k) {
            const float dx = px[k] - cx, dy = py[k] - cy, dz = pz[k] - cz;
            const float d2 = dx * dx + dy * dy + dz * dz;
            md[k] = fminf(md[k], d2);
        }
    }
}

// =====================================================================
// Kernel 2: ball query + K-nearest selection via LDS histogram select.
// one workgroup (256 threads) per center.
// =====================================================================
#define NBINS 2048
#define NCAND 128
__global__ __launch_bounds__(256) void ballq_kernel(
    const float* __restrict__ pos, const float* __restrict__ centers,
    int* __restrict__ nbr, int* __restrict__ cnt)
{
    __shared__ unsigned hist[NBINS];
    __shared__ unsigned chunk[256];
    __shared__ int   list[Kc];
    __shared__ int   cidx[NCAND];
    __shared__ float cd2[NCAND];
    __shared__ int   s_accept, s_cand, s_cut, s_need;

    const int bm = blockIdx.x;
    const int b  = bm / Mc;
    const int t  = threadIdx.x;
    const float* __restrict__ P = pos + (size_t)b * Nc * 3;

    const float cx = centers[(size_t)bm * 3 + 0];
    const float cy = centers[(size_t)bm * 3 + 1];
    const float cz = centers[(size_t)bm * 3 + 2];

    for (int i = t; i < NBINS; i += 256) hist[i] = 0u;
    if (t == 0) { s_accept = 0; s_cand = 0; }
    __syncthreads();

    const float scale = (float)NBINS / R2c;

    // pass 1: histogram of d^2 for in-radius points
    for (int p = t; p < Nc; p += 256) {
        __builtin_prefetch(&P[(p + 512) * 3], 0, 0);
        const float dx = P[p * 3 + 0] - cx;
        const float dy = P[p * 3 + 1] - cy;
        const float dz = P[p * 3 + 2] - cz;
        const float d2 = dx * dx + dy * dy + dz * dz;
        if (d2 <= R2c) {
            int bin = (int)(d2 * scale);
            bin = bin < (NBINS - 1) ? bin : (NBINS - 1);
            atomicAdd(&hist[bin], 1u);
        }
    }
    __syncthreads();

    unsigned s = 0;
#pragma unroll
    for (int j = 0; j < NBINS / 256; ++j) s += hist[t * (NBINS / 256) + j];
    chunk[t] = s;
    __syncthreads();

    if (t == 0) {
        int cut = NBINS, below = 0;
        unsigned cum = 0;
        for (int c = 0; c < 256; ++c) {
            if (cum + chunk[c] >= (unsigned)Kc) {
                unsigned cc = cum;
                for (int j = 0; j < NBINS / 256; ++j) {
                    const unsigned h = hist[c * (NBINS / 256) + j];
                    if (cc + h >= (unsigned)Kc) { cut = c * (NBINS / 256) + j; below = (int)cc; break; }
                    cc += h;
                }
                break;
            }
            cum += chunk[c];
        }
        s_cut  = cut;
        s_need = (cut == NBINS) ? 0 : (Kc - below);
    }
    __syncthreads();
    const int cut = s_cut;

    // pass 2: compact accepted indices; collect boundary-bin candidates
    for (int p = t; p < Nc; p += 256) {
        const float dx = P[p * 3 + 0] - cx;
        const float dy = P[p * 3 + 1] - cy;
        const float dz = P[p * 3 + 2] - cz;
        const float d2 = dx * dx + dy * dy + dz * dz;
        if (d2 <= R2c) {
            int bin = (int)(d2 * scale);
            bin = bin < (NBINS - 1) ? bin : (NBINS - 1);
            if (bin < cut || cut == NBINS) {
                const int slot = atomicAdd(&s_accept, 1);
                if (slot < Kc) list[slot] = p;
            } else if (bin == cut) {
                const int slot = atomicAdd(&s_cand, 1);
                if (slot < NCAND) { cidx[slot] = p; cd2[slot] = d2; }
            }
        }
    }
    __syncthreads();

    if (t == 0) {
        const int nc    = s_cand < NCAND ? s_cand : NCAND;
        int base        = s_accept;
        const int take0 = s_need < nc ? s_need : nc;
        for (int q = 0; q < take0; ++q) {
            int bi = -1; float bd = INFINITY;
            for (int u = 0; u < nc; ++u)
                if (cd2[u] < bd) { bd = cd2[u]; bi = u; }
            list[base + q] = cidx[bi];
            cd2[bi] = INFINITY;
        }
        s_accept = base + take0;
    }
    __syncthreads();

    const int total = s_accept < Kc ? s_accept : Kc;
    if (t < Kc) nbr[(size_t)bm * Kc + t] = (t < total) ? list[t] : -1;
    if (t == 0) cnt[bm] = total;
}

// =====================================================================
// Kernel 3: per-center MLP (WMMA f16, f32 accum) + masked max-pool.
// 128 threads = 4 waves; wave w owns neighbor rows [16w, 16w+16).
//
// All WMMA operands live in LDS PRE-SWIZZLED into the exact per-lane
// register layout (CDNA5 ISA §7.12.2): [tile][lane][16 halves],
// 32 contiguous bytes per lane -> fragment loads are 2x ds_load_b128.
// =====================================================================

// B operand swizzle from global weights:
//   lane 0-15  : col = n0+lane, elem e -> K = k0+(e&7)+(e&8 ? 16 : 0)
//   lane 16-31 : col = n0+lane-16, elem e -> K = k0+8+(e&7)+(e&8 ? 16 : 0)
__device__ __forceinline__ void fill_Bswz(_Float16* dst, const float* __restrict__ W,
                                          int KT, int NT, int ldW, int Kvalid, int t)
{
    const int total = KT * NT * 32 * 16;
    for (int i = t; i < total; i += 128) {
        const int e    = i & 15;
        const int lane = (i >> 4) & 31;
        const int nt   = (i >> 9) % NT;
        const int kt   = (i >> 9) / NT;
        const int k = kt * 32 + ((lane & 16) ? 8 : 0) + (e & 7) + ((e & 8) ? 16 : 0);
        const int n = nt * 16 + (lane & 15);
        dst[i] = (_Float16)((k < Kvalid) ? W[k * ldW + n] : 0.0f);
    }
}

// C/D -> next-layer A-operand scatter (with bias + relu).
// A layout for element (row, k): kk = k&31, kt = k>>5,
//   lane' = (row&15) + 16*((kk>>3)&1),  e = (kk&7) + (kk&16 ? 8 : 0)
// act layout: [(rowtile*KTnext + kt)*32 + lane'] * 16 + e
__device__ __forceinline__ void store_act_swz(
    v8f acc, _Float16* act, int wave, int n0, const float* bias, int lane)
{
    const int col = n0 + (lane & 15);
    const int kt  = col >> 5;
    const int kk  = col & 31;
    const int lhs = (kk >> 3) & 1;
    const int e   = (kk & 7) + ((kk & 16) ? 8 : 0);
    const int hi8 = (lane & 16) ? 8 : 0;
    const float bv = bias[col];
    _Float16* basep = act + (((wave * 2 + kt) * 32) + lhs * 16) * 16 + e;
#pragma unroll
    for (int r = 0; r < 8; ++r) {
        float v = acc[r] + bv;
        v = v > 0.0f ? v : 0.0f;
        basep[(hi8 + r) * 16] = (_Float16)v;
    }
}

#define WMMA_F16(a, b, c) \
    __builtin_amdgcn_wmma_f32_16x16x32_f16(false, (a), false, (b), (short)0, (c), false, false)

__global__ __launch_bounds__(128) void mlp_pool_kernel(
    const float* __restrict__ x,  const float* __restrict__ pos,
    const float* __restrict__ W1, const float* __restrict__ b1,
    const float* __restrict__ W2, const float* __restrict__ b2,
    const float* __restrict__ W3, const float* __restrict__ b3,
    const int* __restrict__ nbr,  const int* __restrict__ cnt,
    const float* __restrict__ centers, float* __restrict__ out)
{
    // pre-swizzled WMMA operands, [tile][lane][16 halves]
    __shared__ __align__(32) _Float16 w1s[1 * 4 * 32 * 16];   // K padded 6->32
    __shared__ __align__(32) _Float16 w2s[2 * 4 * 32 * 16];
    __shared__ __align__(32) _Float16 w3s[2 * 8 * 32 * 16];
    __shared__ __align__(32) _Float16 fsw[4 * 1 * 32 * 16];   // layer-1 A (K=32)
    __shared__ __align__(32) _Float16 actA[4 * 2 * 32 * 16];  // layer-1 out / layer-2 A
    __shared__ __align__(32) _Float16 actB[4 * 2 * 32 * 16];  // layer-2 out / layer-3 A
    __shared__ float bias1[H1c], bias2[H2c], bias3[HOc];
    __shared__ float pmax[4][HOc];

    const int bm   = blockIdx.x;
    const int b    = bm / Mc;
    const int t    = threadIdx.x;
    const int lane = t & 31;
    const int wave = t >> 5;

    // ---- stage weights (pre-swizzled) / biases ----
    fill_Bswz(w1s, W1, 1, 4, H1c, Cc + 3, t);
    fill_Bswz(w2s, W2, 2, 4, H2c, H1c,     t);
    fill_Bswz(w3s, W3, 2, 8, HOc, H2c,     t);
    if (t < H1c) { bias1[t] = b1[t]; bias2[t] = b2[t]; }
    if (t < HOc) bias3[t] = b3[t];

    // ---- gather features straight into A-operand layout ----
    const float cx = centers[(size_t)bm * 3 + 0];
    const float cy = centers[(size_t)bm * 3 + 1];
    const float cz = centers[(size_t)bm * 3 + 2];
    const int count = cnt[bm];

    if (t < Kc) {
        const int j  = nbr[(size_t)bm * Kc + t];
        const int rt = t >> 4, rl = t & 15;
        v16h frow = {};
        if (j >= 0) {
            const float* xp = x   + ((size_t)b * Nc + j) * Cc;
            const float* pp = pos + ((size_t)b * Nc + j) * 3;
            frow[0] = (_Float16)xp[0];
            frow[1] = (_Float16)xp[1];
            frow[2] = (_Float16)xp[2];
            frow[3] = (_Float16)(pp[0] - cx);
            frow[4] = (_Float16)(pp[1] - cy);
            frow[5] = (_Float16)(pp[2] - cz);
        }
        v16h z = {};
        *(v16h*)(fsw + (rt * 32 + rl)      * 16) = frow;  // K 0..7 / 16..23 group
        *(v16h*)(fsw + (rt * 32 + rl + 16) * 16) = z;     // K 8..15 / 24..31 group
    }
    __syncthreads();

    // ---- layer 1: [64x32] x [32x64] -> actA ----
    {
        const v16h a = *(const v16h*)(fsw + (wave * 32 + lane) * 16);
#pragma unroll
        for (int nt = 0; nt < H1c / 16; ++nt) {
            v8f c = {};
            const v16h bmat = *(const v16h*)(w1s + (nt * 32 + lane) * 16);
            c = WMMA_F16(a, bmat, c);
            store_act_swz(c, actA, wave, nt * 16, bias1, lane);
        }
    }
    __syncthreads();

    // ---- layer 2: [64x64] x [64x64] -> actB ----
    {
        const v16h a0 = *(const v16h*)(actA + ((wave * 2 + 0) * 32 + lane) * 16);
        const v16h a1 = *(const v16h*)(actA + ((wave * 2 + 1) * 32 + lane) * 16);
#pragma unroll
        for (int nt = 0; nt < H2c / 16; ++nt) {
            v8f c = {};
            const v16h b0m = *(const v16h*)(w2s + ((0 * 4 + nt) * 32 + lane) * 16);
            const v16h b1m = *(const v16h*)(w2s + ((1 * 4 + nt) * 32 + lane) * 16);
            c = WMMA_F16(a0, b0m, c);
            c = WMMA_F16(a1, b1m, c);
            store_act_swz(c, actB, wave, nt * 16, bias2, lane);
        }
    }
    __syncthreads();

    // ---- layer 3: [64x64] x [64x128], bias (no relu), masked max over rows ----
    {
        const v16h a0 = *(const v16h*)(actB + ((wave * 2 + 0) * 32 + lane) * 16);
        const v16h a1 = *(const v16h*)(actB + ((wave * 2 + 1) * 32 + lane) * 16);
        const int m0 = wave * 16;
#pragma unroll
        for (int nt = 0; nt < HOc / 16; ++nt) {
            v8f c = {};
            const v16h b0m = *(const v16h*)(w3s + ((0 * 8 + nt) * 32 + lane) * 16);
            const v16h b1m = *(const v16h*)(w3s + ((1 * 8 + nt) * 32 + lane) * 16);
            c = WMMA_F16(a0, b0m, c);
            c = WMMA_F16(a1, b1m, c);
            const int col = nt * 16 + (lane & 15);
            const int rb  = m0 + ((lane & 16) ? 8 : 0);
            const float bv = bias3[col];
            float best = -INFINITY;
#pragma unroll
            for (int r = 0; r < 8; ++r)
                if (rb + r < count) best = fmaxf(best, c[r] + bv);
            const float other = __shfl_xor(best, 16);
            best = fmaxf(best, other);
            if (lane < 16) pmax[wave][col] = best;
        }
    }
    __syncthreads();

    if (t < HOc) {
        float m = fmaxf(fmaxf(pmax[0][t], pmax[1][t]), fmaxf(pmax[2][t], pmax[3][t]));
        out[(size_t)bm * HOc + t] = (m > -1e37f) ? m : 0.0f;   // empty-neighborhood guard
    }
}

// =====================================================================
// launch
// =====================================================================
extern "C" void kernel_launch(void* const* d_in, const int* in_sizes, int n_in,
                              void* d_out, int out_size, void* d_ws, size_t ws_size,
                              hipStream_t stream)
{
    (void)in_sizes; (void)n_in; (void)out_size; (void)ws_size;
    const float* x   = (const float*)d_in[0];
    const float* pos = (const float*)d_in[1];
    const float* W1  = (const float*)d_in[2];
    const float* b1  = (const float*)d_in[3];
    const float* W2  = (const float*)d_in[4];
    const float* b2  = (const float*)d_in[5];
    const float* W3  = (const float*)d_in[6];
    const float* b3  = (const float*)d_in[7];

    float* out     = (float*)d_out;
    float* centers = out + (size_t)Bc * Mc * HOc;   // second tuple element

    int* idx = (int*)d_ws;
    int* nbr = idx + Bc * Mc;
    int* cnt = nbr + (size_t)Bc * Mc * Kc;

    fps_kernel<<<Bc, 1024, 0, stream>>>(pos, idx, centers);
    ballq_kernel<<<Bc * Mc, 256, 0, stream>>>(pos, centers, nbr, cnt);
    mlp_pool_kernel<<<Bc * Mc, 128, 0, stream>>>(x, pos, W1, b1, W2, b2, W3, b3,
                                                 nbr, cnt, centers, out);
}
